// SelfAttention_25752623907480
// MI455X (gfx1250) — compile-verified, matrix operand order checked by reading, not run on hardware
//
#include <hip/hip_runtime.h>
#include <hip/hip_bf16.h>

// ---------------------------------------------------------------------------
// Self-attention forward for MI455X (gfx1250, wave32, WMMA 16x16x32 f16)
//   B=4, T=2048, C=1024, H=16, D=64
// ---------------------------------------------------------------------------

typedef _Float16 v16h  __attribute__((ext_vector_type(16)));
typedef _Float16 half8 __attribute__((ext_vector_type(8)));
typedef _Float16 half4 __attribute__((ext_vector_type(4)));
typedef float    v8f   __attribute__((ext_vector_type(8)));

typedef unsigned int u32x4 __attribute__((ext_vector_type(4)));
typedef int          i32x8 __attribute__((ext_vector_type(8)));
typedef int          i32x4 __attribute__((ext_vector_type(4)));

#define NEG_INF (-__builtin_inff())

#if defined(__AMDGCN__) && __has_builtin(__builtin_amdgcn_tensor_load_to_lds) && \
    __has_builtin(__builtin_amdgcn_s_wait_tensorcnt)
#define USE_TDM 1
#else
#define USE_TDM 0
#endif

static constexpr int Bsz = 4;
static constexpr int T   = 2048;
static constexpr int C   = 1024;
static constexpr int H   = 16;
static constexpr int D   = 64;
static constexpr int M   = Bsz * T;   // 8192 rows

__device__ __forceinline__ v8f wmma_f16(v16h a, v16h b, v8f c) {
  return __builtin_amdgcn_wmma_f32_16x16x32_f16(
      /*neg_a=*/false, a, /*neg_b=*/false, b,
      /*c_mod=*/(short)0, c, /*reuse_a=*/false, /*reuse_b=*/false);
}

// Build a 16-half fragment from two contiguous 8-half (16B) chunks.
__device__ __forceinline__ v16h mk16(const _Float16* lo, const _Float16* hi) {
  half8 a = *(const half8*)lo;
  half8 b = *(const half8*)hi;
  v16h r;
#pragma unroll
  for (int i = 0; i < 8; ++i) { r[i] = a[i]; r[i + 8] = b[i]; }
  return r;
}

#if USE_TDM
// Issue a TDM 2D tile load: global (rows x tileW halves, row stride =
// strideElems) -> LDS at ldsAddr, packed row-major. D# per ISA 08 §8.
// This toolchain exposes the 6-arg builtin: (g0, g1, g2, g3, g4, cpol).
__device__ __forceinline__ void tdm_load_tile_f16(const _Float16* gptr,
                                                  unsigned ldsAddr,
                                                  unsigned tileW,   // halves
                                                  unsigned tileH,   // rows
                                                  unsigned strideElems) {
  unsigned long long ga = (unsigned long long)(size_t)gptr;
  u32x4 g0;
  g0[0] = 1u;                                   // count=1, user mode
  g0[1] = ldsAddr;                              // lds_addr (bytes)
  g0[2] = (unsigned)(ga & 0xFFFFFFFFu);         // global_addr[31:0]
  g0[3] = (unsigned)((ga >> 32) & 0x01FFFFFFu)  // global_addr[56:32]
        | (2u << 30);                           // type=2 ("image")
  i32x8 g1;
  unsigned td0 = tileW, td1 = tileH;            // tensor dims = tile (no OOB)
  unsigned long long s0 = strideElems;          // dim0 stride (elements)
  g1[0] = (int)(1u << 16);                      // data_size=1 -> 2 bytes
  g1[1] = (int)((td0 & 0xFFFFu) << 16);         // tensor_dim0[15:0]
  g1[2] = (int)(((td0 >> 16) & 0xFFFFu) | ((td1 & 0xFFFFu) << 16));
  g1[3] = (int)(((td1 >> 16) & 0xFFFFu) | (tileW << 16));   // tile_dim0
  g1[4] = (int)tileH;                           // tile_dim1 (tile_dim2=0)
  g1[5] = (int)(unsigned)(s0 & 0xFFFFFFFFu);    // tensor_dim0_stride[31:0]
  g1[6] = (int)(unsigned)((s0 >> 32) & 0xFFFFu);
  g1[7] = 0;
  i32x4 g2 = {0, 0, 0, 0};                      // 2D tensor: groups 2/3 unused
  i32x4 g3 = {0, 0, 0, 0};
  i32x8 g4 = {0, 0, 0, 0, 0, 0, 0, 0};
  __builtin_amdgcn_tensor_load_to_lds(g0, g1, g2, g3, g4, 0);
}
#endif

// ---------------------------------------------------------------------------
// f32 -> f16 conversion (vectorized x4)
// ---------------------------------------------------------------------------
__global__ void cvt_f32_to_f16(const float* __restrict__ in,
                               _Float16* __restrict__ out, int n4) {
  int i = blockIdx.x * blockDim.x + threadIdx.x;
  if (i < n4) {
    float4 f = ((const float4*)in)[i];
    half4 h;
    h[0] = (_Float16)f.x; h[1] = (_Float16)f.y;
    h[2] = (_Float16)f.z; h[3] = (_Float16)f.w;
    ((half4*)out)[i] = h;
  }
}

// ---------------------------------------------------------------------------
// WMMA GEMM core: 256 threads = 8 waves; WG computes 64(M) x 128(N).
// Each wave: one A fragment reused over a 16x64 strip = 4 WMMAs per k-step.
// A tile staged by the Tensor Data Mover (async DMA, TENSORcnt); B tile
// transposed through VGPRs (TDM cannot transpose).
// ---------------------------------------------------------------------------
struct EpiQKV {
  _Float16* q; _Float16* k; _Float16* v;
  const float* bias;
  __device__ void operator()(int row, int col, float acc) const {
    float val = acc + bias[col];
    int which = col >> 10;        // 0=q 1=k 2=v
    int c     = col & 1023;
    int hh    = c >> 6;
    int d     = c & 63;
    int b     = row >> 11;
    int t     = row & 2047;
    if (which == 0) val *= 0.125f;          // fold 1/sqrt(64) into Q
    _Float16* dst = (which == 0) ? q : ((which == 1) ? k : v);
    dst[(((size_t)(b * H + hh)) * T + t) * D + d] = (_Float16)val;
  }
};

struct EpiProj {
  float* out; const float* bias;
  __device__ void operator()(int row, int col, float acc) const {
    out[(size_t)row * C + col] = acc + bias[col];
  }
};

template <typename Epi>
__global__ __launch_bounds__(256)
void gemm_wmma(const _Float16* __restrict__ A, const _Float16* __restrict__ B,
               int N, int K, Epi epi) {
  __shared__ _Float16 As[64 * 32];      // 64 rows (M) x 32 (K)
  __shared__ _Float16 Bts[128 * 32];    // transposed: 128 cols (N) x 32 (K)

  const int tid   = threadIdx.x;
  const int lane  = tid & 31;
  const int wid   = tid >> 5;
  const int tileM = blockIdx.y * 64;
  const int tileN = blockIdx.x * 128;
  const int wm    = (wid >> 1) * 16;    // 0..48
  const int wn    = (wid & 1) * 64;     // 0 or 64
  const int lm    = lane & 15;
  const int ksel  = (lane >> 4) & 1;

  v8f acc[4] = {};

  for (int k0 = 0; k0 < K; k0 += 32) {
#if USE_TDM
    // Async A-tile DMA: 64 rows x 32 halves, issued once by wave 0.
    // Previous iteration's trailing barrier guarantees As is free.
    if (wid == 0) {
      tdm_load_tile_f16(&A[(size_t)tileM * K + k0],
                        (unsigned)(size_t)&As[0], 32u, 64u, (unsigned)K);
    }
#else
    {
      int row = tid >> 2;
      int cs  = (tid & 3) * 8;
      *(half8*)(&As[row * 32 + cs]) =
          *(const half8*)(&A[(size_t)(tileM + row) * K + k0 + cs]);
    }
#endif
    // Stage B tile transposed: each thread loads 16 halves of one B row,
    // scatters into K-contiguous layout.
    {
      int kr = tid >> 3;
      int ns = (tid & 7) * 16;
      const _Float16* bp = &B[(size_t)(k0 + kr) * N + tileN + ns];
      half8 b0 = *(const half8*)bp;
      half8 b1 = *(const half8*)(bp + 8);
#pragma unroll
      for (int j = 0; j < 8; ++j) {
        Bts[(ns + j) * 32 + kr]     = b0[j];
        Bts[(ns + 8 + j) * 32 + kr] = b1[j];
      }
    }
#if USE_TDM
    if (wid == 0) __builtin_amdgcn_s_wait_tensorcnt(0);  // DMA done before barrier
#endif
    __syncthreads();

    const _Float16* ap = &As[(wm + lm) * 32];
    v16h af = mk16(ap + 8 * ksel, ap + 16 + 8 * ksel);
#pragma unroll
    for (int nt = 0; nt < 4; ++nt) {
      const _Float16* bp = &Bts[(wn + nt * 16 + lm) * 32];
      v16h bf = mk16(bp + 8 * ksel, bp + 16 + 8 * ksel);
      acc[nt] = wmma_f16(af, bf, acc[nt]);
    }
    __syncthreads();
  }

#pragma unroll
  for (int nt = 0; nt < 4; ++nt) {
#pragma unroll
    for (int i = 0; i < 8; ++i) {
      int m = tileM + wm + i + 8 * ksel;   // C-layout: VGPR i -> M=i / M=8+i
      int n = tileN + wn + nt * 16 + lm;
      epi(m, n, acc[nt][i]);
    }
  }
}

// ---------------------------------------------------------------------------
// Flash attention: Q,K,V f16 [B,H,T,D] (Q pre-scaled). Output y f16 [B,T,C].
// 128 threads = 4 waves; each wave owns a 16-query tile; WG covers 64 queries.
// Key loop in blocks of 64 with cooperative LDS transpose of V.
// Per 64-key block per wave: 8 WMMAs for S, 8 WMMAs for P.V.
// ---------------------------------------------------------------------------
__global__ __launch_bounds__(128)
void flash_attn(const _Float16* __restrict__ q, const _Float16* __restrict__ k,
                const _Float16* __restrict__ v, _Float16* __restrict__ y) {
  __shared__ _Float16 Vt[64 * 64];        // transposed V block: [d][key]
  __shared__ _Float16 Pb[4 * 16 * 64];    // per-wave P tile (16 q x 64 key)

  const int tid  = threadIdx.x;
  const int lane = tid & 31;
  const int wid  = tid >> 5;
  const int b    = blockIdx.z;
  const int h    = blockIdx.y;
  const int q0   = blockIdx.x * 64;
  const int qt   = q0 + wid * 16;
  const int lm   = lane & 15;
  const int ksel = (lane >> 4) & 1;

  const size_t base = ((size_t)(b * H + h)) * T * D;
  const _Float16* qp = q + base;
  const _Float16* kp = k + base;
  const _Float16* vp = v + base;

  // Q fragments (A-layout): lane lm holds row qt+lm, 16B-contiguous K runs.
  const _Float16* qrow = qp + (size_t)(qt + lm) * D;
  v16h Qa0 = mk16(qrow + 8 * ksel,      qrow + 16 + 8 * ksel);
  v16h Qa1 = mk16(qrow + 32 + 8 * ksel, qrow + 48 + 8 * ksel);

  v8f O[4] = {};
  float mx[8], ls[8];
#pragma unroll
  for (int i = 0; i < 8; ++i) { mx[i] = NEG_INF; ls[i] = 0.0f; }

  _Float16* Pw = &Pb[wid * 16 * 64];
  const int nblocks = (q0 >> 6) + 1;      // 64-key blocks up to q0+63

  for (int blk = 0; blk < nblocks; ++blk) {
    const int jb = blk * 64;

    __syncthreads();   // protect previous iteration's Vt reads
    {  // cooperative stage: Vt[d][key] <- V[jb+key][d]; 128 thr x 32 halves
      int kr = tid >> 1;            // key 0..63
      int ds = (tid & 1) * 32;      // d segment
      const _Float16* vrow = vp + (size_t)(jb + kr) * D + ds;
      half8 v0 = *(const half8*)vrow;
      half8 v1 = *(const half8*)(vrow + 8);
      half8 v2 = *(const half8*)(vrow + 16);
      half8 v3 = *(const half8*)(vrow + 24);
#pragma unroll
      for (int j = 0; j < 8; ++j) {
        Vt[(ds + j) * 64 + kr]      = v0[j];
        Vt[(ds + 8 + j) * 64 + kr]  = v1[j];
        Vt[(ds + 16 + j) * 64 + kr] = v2[j];
        Vt[(ds + 24 + j) * 64 + kr] = v3[j];
      }
    }
    __syncthreads();

    if (jb <= qt + 15) {            // wave-uniform: EXEC stays all-ones
      // S = Q.K^T over four 16-key sub-tiles
      v8f S[4] = {};
#pragma unroll
      for (int st = 0; st < 4; ++st) {
        const _Float16* krow = kp + (size_t)(jb + st * 16 + lm) * D;
        v16h Klo = mk16(krow + 8 * ksel,      krow + 16 + 8 * ksel);
        v16h Khi = mk16(krow + 32 + 8 * ksel, krow + 48 + 8 * ksel);
        S[st] = wmma_f16(Qa0, Klo, S[st]);
        S[st] = wmma_f16(Qa1, Khi, S[st]);
      }

      const bool needMask = (jb + 63) > qt;

#pragma unroll
      for (int i = 0; i < 8; ++i) {
        const int qi = qt + i + 8 * ksel;          // this element's query row
        float s0 = S[0][i], s1 = S[1][i], s2 = S[2][i], s3 = S[3][i];
        if (needMask) {
          s0 = (jb + lm      > qi) ? NEG_INF : s0; // per-lane select, no EXEC
          s1 = (jb + 16 + lm > qi) ? NEG_INF : s1;
          s2 = (jb + 32 + lm > qi) ? NEG_INF : s2;
          s3 = (jb + 48 + lm > qi) ? NEG_INF : s3;
        }
        // row max across the 16 lanes holding this row (half-wave shuffle)
        float bm = fmaxf(fmaxf(s0, s1), fmaxf(s2, s3));
#pragma unroll
        for (int off = 1; off < 16; off <<= 1)
          bm = fmaxf(bm, __shfl_xor(bm, off, 16));
        const float mnew = fmaxf(mx[i], bm);
        const float f    = __expf(mx[i] - mnew);
        const float p0   = __expf(s0 - mnew);
        const float p1   = __expf(s1 - mnew);
        const float p2   = __expf(s2 - mnew);
        const float p3   = __expf(s3 - mnew);
        float rs = (p0 + p1) + (p2 + p3);
#pragma unroll
        for (int off = 1; off < 16; off <<= 1)
          rs += __shfl_xor(rs, off, 16);
        ls[i] = ls[i] * f + rs;
        mx[i] = mnew;
        O[0][i] *= f; O[1][i] *= f; O[2][i] *= f; O[3][i] *= f;
        // scatter P into per-wave LDS tile (row-major 16x64)
        const int m = i + 8 * ksel;
        Pw[m * 64 + lm]      = (_Float16)p0;
        Pw[m * 64 + 16 + lm] = (_Float16)p1;
        Pw[m * 64 + 32 + lm] = (_Float16)p2;
        Pw[m * 64 + 48 + lm] = (_Float16)p3;
      }

      // same-wave LDS RAW: DS ops are in-order within a wave
      const _Float16* pr = &Pw[lm * 64];
      v16h Pa0 = mk16(pr + 8 * ksel,      pr + 16 + 8 * ksel);   // keys 0..31
      v16h Pa1 = mk16(pr + 32 + 8 * ksel, pr + 48 + 8 * ksel);   // keys 32..63

#pragma unroll
      for (int nt = 0; nt < 4; ++nt) {
        const _Float16* vr = &Vt[(nt * 16 + lm) * 64];
        v16h Vb0 = mk16(vr + 8 * ksel,      vr + 16 + 8 * ksel);
        v16h Vb1 = mk16(vr + 32 + 8 * ksel, vr + 48 + 8 * ksel);
        O[nt] = wmma_f16(Pa0, Vb0, O[nt]);
        O[nt] = wmma_f16(Pa1, Vb1, O[nt]);
      }
    }
  }

  // finalize: y[b, qt+m, h*64 + d]  (row-major [B,T,C] for the proj GEMM)
#pragma unroll
  for (int i = 0; i < 8; ++i) {
    const float inv = 1.0f / ls[i];
    const int m = i + 8 * ksel;
    const size_t rowoff = ((size_t)b * T + qt + m) * C + h * D;
    y[rowoff + 0  + lm] = (_Float16)(O[0][i] * inv);
    y[rowoff + 16 + lm] = (_Float16)(O[1][i] * inv);
    y[rowoff + 32 + lm] = (_Float16)(O[2][i] * inv);
    y[rowoff + 48 + lm] = (_Float16)(O[3][i] * inv);
  }
}

// ---------------------------------------------------------------------------
// Host launcher
// ---------------------------------------------------------------------------
extern "C" void kernel_launch(void* const* d_in, const int* in_sizes, int n_in,
                              void* d_out, int out_size, void* d_ws, size_t ws_size,
                              hipStream_t stream) {
  const float* x      = (const float*)d_in[0];   // [B,T,C]
  const float* w_qkv  = (const float*)d_in[1];   // [C,3C]
  const float* b_qkv  = (const float*)d_in[2];   // [3C]
  const float* w_proj = (const float*)d_in[3];   // [C,C]
  const float* b_proj = (const float*)d_in[4];   // [C]
  float* out = (float*)d_out;

  // Workspace layout (f16 buffers, 256B aligned)
  char* ws = (char*)d_ws;
  size_t off = 0;
  auto alloc = [&](size_t bytes) {
    void* p = ws + off;
    off = (off + bytes + 255) & ~(size_t)255;
    return p;
  };
  _Float16* x16  = (_Float16*)alloc((size_t)M * C * 2);
  _Float16* wq16 = (_Float16*)alloc((size_t)C * 3 * C * 2);
  _Float16* wp16 = (_Float16*)alloc((size_t)C * C * 2);
  _Float16* q16  = (_Float16*)alloc((size_t)M * C * 2);
  _Float16* k16  = (_Float16*)alloc((size_t)M * C * 2);
  _Float16* v16  = (_Float16*)alloc((size_t)M * C * 2);
  _Float16* y16  = (_Float16*)alloc((size_t)M * C * 2);
  (void)ws_size;

  // 1) convert inputs to f16
  {
    int n4 = (M * C) / 4;
    cvt_f32_to_f16<<<(n4 + 255) / 256, 256, 0, stream>>>(x, x16, n4);
    n4 = (C * 3 * C) / 4;
    cvt_f32_to_f16<<<(n4 + 255) / 256, 256, 0, stream>>>(w_qkv, wq16, n4);
    n4 = (C * C) / 4;
    cvt_f32_to_f16<<<(n4 + 255) / 256, 256, 0, stream>>>(w_proj, wp16, n4);
  }

  // 2) QKV projection GEMM with scatter epilogue (Q pre-scaled by 1/8)
  {
    EpiQKV epi{q16, k16, v16, b_qkv};
    dim3 grid((3 * C) / 128, M / 64);
    gemm_wmma<EpiQKV><<<grid, 256, 0, stream>>>(x16, wq16, 3 * C, C, epi);
  }

  // 3) causal flash attention
  {
    dim3 grid(T / 64, H, Bsz);
    flash_attn<<<grid, 128, 0, stream>>>(q16, k16, v16, y16);
  }

  // 4) output projection GEMM
  {
    EpiProj epi{out, b_proj};
    dim3 grid(C / 128, M / 64);
    gemm_wmma<EpiProj><<<grid, 256, 0, stream>>>(y16, wp16, C, C, epi);
  }
}
